// OutputKAN_2594160247631
// MI455X (gfx1250) — compile-verified, matrix operand order checked by reading, not run on hardware
//
#include <hip/hip_runtime.h>
#include <hip/hip_bf16.h>

// ---------------- problem constants ----------------
#define Bn    512
#define Tn    197
#define Cn    768
#define On    1000
#define Gn    8
#define KTOT  6912          // Cn + Cn*Gn
#define NPAD  1024          // On padded to multiple of 128
#define NTC   (Tn * Cn)     // 151296

typedef __attribute__((ext_vector_type(16))) __bf16 v16bf;
typedef __attribute__((ext_vector_type(8)))  float  v8f;

union Frag { v16bf v; uint4 q[2]; };

// ---------------- workspace layout (bytes) ----------------
// [0,     3072)   Wc  : sum_t ln_weight  (768 f32)
// [4096,  7168)   Bc  : sum_t ln_bias    (768 f32)
// [8192, +A)      A   : 512 x 6912 bf16  (silu(pooled) | spline bases)
// [.., +B)        Bm  : 1024 x 6912 bf16 (base_weight | scaled spline w, zero-pad rows)
#define WS_WC 0
#define WS_BC 4096
#define WS_A  8192
#define WS_B  (8192 + (size_t)Bn * KTOT * 2)

// ---------------- cubic B-spline bases (uniform grid, h = 0.4) ----------------
__device__ __forceinline__ void bsplines8(float v, float* outb) {
    float b[11];
#pragma unroll
    for (int j = 0; j < 11; ++j) {
        const float g0 = 0.4f * (float)j - 2.2f;
        const float g1 = g0 + 0.4f;
        b[j] = (v >= g0 && v < g1) ? 1.0f : 0.0f;
    }
#pragma unroll
    for (int k = 1; k <= 3; ++k) {
        const float inv = 1.0f / (0.4f * (float)k);
#pragma unroll
        for (int j = 0; j < 10; ++j) {
            if (j < 11 - k) {
                const float gj   = 0.4f * (float)j - 2.2f;
                const float gjk1 = 0.4f * (float)(j + k + 1) - 2.2f;
                b[j] = (v - gj) * inv * b[j] + (gjk1 - v) * inv * b[j + 1];
            }
        }
    }
#pragma unroll
    for (int g = 0; g < 8; ++g) outb[g] = b[g];
}

// ---------------- kernel 0: reduce LN params over tokens ----------------
__global__ void kan_lnreduce(const float* __restrict__ lnw,
                             const float* __restrict__ lnb,
                             float* __restrict__ Wc, float* __restrict__ Bc) {
    const int c = blockIdx.x * 128 + threadIdx.x;
    if (c < Cn) {
        float sw = 0.f, sb = 0.f;
        for (int t = 0; t < Tn; ++t) {
            sw += lnw[t * Cn + c];
            sb += lnb[t * Cn + c];
        }
        Wc[c] = sw;
        Bc[c] = sb;
    }
}

// ---------------- kernel 1: build bf16 B matrix (N-major, K contiguous) ----------------
__global__ __launch_bounds__(256) void kan_prepB(const float* __restrict__ bw,
                                                 const float* __restrict__ sw,
                                                 const float* __restrict__ ss,
                                                 __bf16* __restrict__ Bm) {
    const int id = blockIdx.x * 256 + threadIdx.x;   // 0 .. NPAD*Cn-1
    const int o = id / Cn;
    const int c = id % Cn;
    __bf16* row = Bm + (size_t)o * KTOT;
    if (o < On) {
        row[c] = (__bf16)bw[o * Cn + c];
        const float sc = ss[o * Cn + c];
        const float* w = sw + ((size_t)o * Cn + c) * Gn;
#pragma unroll
        for (int g = 0; g < Gn; ++g)
            row[Cn + c * Gn + g] = (__bf16)(w[g] * sc);
    } else {
        row[c] = (__bf16)0.0f;
#pragma unroll
        for (int g = 0; g < Gn; ++g)
            row[Cn + c * Gn + g] = (__bf16)0.0f;
    }
}

// ---------------- kernel 2: fused LN stats + weighted token-pool + spline A build ----------------
// One workgroup (256 threads) per batch sample; x is read exactly once.
__global__ __launch_bounds__(256) void kan_pass1(const float* __restrict__ x,
                                                 const float* __restrict__ lnw,
                                                 const float* __restrict__ Wc,
                                                 const float* __restrict__ Bc,
                                                 __bf16* __restrict__ A) {
    const int b   = blockIdx.x;
    const int tid = threadIdx.x;
    const float* xb = x + (size_t)b * NTC;

    float s1[3] = {0.f, 0.f, 0.f};
    float lsum = 0.f, lsq = 0.f;

    for (int t = 0; t < Tn; ++t) {
        const float* xr = xb  + t * Cn;
        const float* wr = lnw + t * Cn;
#pragma unroll
        for (int j = 0; j < 3; ++j) {
            const int c = tid + j * 256;
            const float xv = xr[c];
            const float wv = wr[c];
            lsum += xv;
            lsq  += xv * xv;
            s1[j] += xv * wv;
        }
    }

    __shared__ float sh[512];
    sh[tid]       = lsum;
    sh[tid + 256] = lsq;
    __syncthreads();
#pragma unroll
    for (int s = 128; s > 0; s >>= 1) {
        if (tid < s) {
            sh[tid]       += sh[tid + s];
            sh[tid + 256] += sh[tid + 256 + s];
        }
        __syncthreads();
    }
    if (tid == 0) {
        const float inv_n = 1.0f / (float)NTC;
        const float mean  = sh[0] * inv_n;
        const float var   = sh[256] * inv_n - mean * mean;
        sh[0] = mean;
        sh[1] = rsqrtf(var + 1e-5f);
    }
    __syncthreads();
    const float mean = sh[0];
    const float rstd = sh[1];
    const float invT = 1.0f / (float)Tn;

    __bf16* Arow = A + (size_t)b * KTOT;
#pragma unroll
    for (int j = 0; j < 3; ++j) {
        const int c = tid + j * 256;
        const float pooled = (rstd * (s1[j] - mean * Wc[c]) + Bc[c]) * invT;
        // base path: silu
        const float sil = pooled / (1.0f + __expf(-pooled));
        Arow[c] = (__bf16)sil;
        // spline path: 8 cubic bases
        float bb[8];
        bsplines8(pooled, bb);
        __bf16* sp = Arow + Cn + c * Gn;
#pragma unroll
        for (int g = 0; g < 8; ++g) sp[g] = (__bf16)bb[g];
    }
}

// ---------------- kernel 3: WMMA GEMM  out[512x1000] = A[512xK] * Bm[1024xK]^T ----------------
// Workgroup: 256 threads = 8 waves in a 2x4 layout; wave tile 32x32 = 2x2 WMMA accums.
// Operands are L2-resident (A 6.9MB + B 13.8MB << 192MB L2): direct global fragment loads.
__global__ __launch_bounds__(256) void kan_gemm(const __bf16* __restrict__ A,
                                                const __bf16* __restrict__ Bm,
                                                float* __restrict__ out) {
    const int lane = threadIdx.x & 31;
    const int w    = threadIdx.x >> 5;
    const int wr   = w >> 2;          // 0..1  (M)
    const int wc   = w & 3;           // 0..3  (N)
    const int m0   = blockIdx.y * 64  + wr * 32;
    const int n0   = blockIdx.x * 128 + wc * 32;

    // gfx1250 16-bit A/B fragment layout:
    //   lane L<16 : row (L&15), K chunks {0..7} and {16..23}
    //   lane L>=16: row (L&15), K chunks {8..15} and {24..31}
    const int rA = lane & 15;
    const int kh = (lane & 16) ? 8 : 0;

    const __bf16* a0 = A  + (size_t)(m0 + rA) * KTOT + kh;
    const __bf16* a1 = a0 + (size_t)16 * KTOT;
    const __bf16* b0 = Bm + (size_t)(n0 + rA) * KTOT + kh;
    const __bf16* b1 = b0 + (size_t)16 * KTOT;

    v8f acc00 = {}, acc01 = {}, acc10 = {}, acc11 = {};

    for (int k = 0; k < KTOT; k += 32) {
        if (k + 256 < KTOT) {
            __builtin_prefetch(a0 + k + 256, 0, 3);   // global_prefetch_b8
            __builtin_prefetch(b0 + k + 256, 0, 3);
        }
        Frag fa0, fa1, fb0, fb1;
        fa0.q[0] = *(const uint4*)(a0 + k);
        fa0.q[1] = *(const uint4*)(a0 + k + 16);
        fa1.q[0] = *(const uint4*)(a1 + k);
        fa1.q[1] = *(const uint4*)(a1 + k + 16);
        fb0.q[0] = *(const uint4*)(b0 + k);
        fb0.q[1] = *(const uint4*)(b0 + k + 16);
        fb1.q[0] = *(const uint4*)(b1 + k);
        fb1.q[1] = *(const uint4*)(b1 + k + 16);

        acc00 = __builtin_amdgcn_wmma_f32_16x16x32_bf16(false, fa0.v, false, fb0.v,
                                                        (short)0, acc00, false, false);
        acc01 = __builtin_amdgcn_wmma_f32_16x16x32_bf16(false, fa0.v, false, fb1.v,
                                                        (short)0, acc01, false, false);
        acc10 = __builtin_amdgcn_wmma_f32_16x16x32_bf16(false, fa1.v, false, fb0.v,
                                                        (short)0, acc10, false, false);
        acc11 = __builtin_amdgcn_wmma_f32_16x16x32_bf16(false, fa1.v, false, fb1.v,
                                                        (short)0, acc11, false, false);
    }

    // f32 C/D layout: VGPR r, lanes 0-15 -> M = r, lanes 16-31 -> M = r + 8; N = lane&15
    const int nc = lane & 15;
    const int mr = (lane & 16) ? 8 : 0;
#pragma unroll
    for (int r = 0; r < 8; ++r) {
        const int m  = m0 + mr + r;
        const int m1 = m + 16;
        const int nA = n0 + nc;
        const int nB = n0 + 16 + nc;
        if (nA < On) {
            out[(size_t)m  * On + nA] = acc00[r];
            out[(size_t)m1 * On + nA] = acc10[r];
        }
        if (nB < On) {
            out[(size_t)m  * On + nB] = acc01[r];
            out[(size_t)m1 * On + nB] = acc11[r];
        }
    }
}

// ---------------- launcher ----------------
extern "C" void kernel_launch(void* const* d_in, const int* in_sizes, int n_in,
                              void* d_out, int out_size, void* d_ws, size_t ws_size,
                              hipStream_t stream) {
    const float* x   = (const float*)d_in[0];   // (512,197,768)
    const float* lnw = (const float*)d_in[1];   // (197,768)
    const float* lnb = (const float*)d_in[2];   // (197,768)
    const float* bw  = (const float*)d_in[3];   // (1000,768)
    const float* sw  = (const float*)d_in[4];   // (1000,768,8)
    const float* ss  = (const float*)d_in[5];   // (1000,768)
    float* out = (float*)d_out;                 // (512,1000)

    char* ws = (char*)d_ws;
    float*  Wc = (float*)(ws + WS_WC);
    float*  Bc = (float*)(ws + WS_BC);
    __bf16* A  = (__bf16*)(ws + WS_A);
    __bf16* Bm = (__bf16*)(ws + WS_B);

    // 0) LN param token-sums (tiny)
    kan_lnreduce<<<6, 128, 0, stream>>>(lnw, lnb, Wc, Bc);
    // 1) bf16 B matrix: [base_weight | spline_weight*scaler], zero-padded to 1024 rows
    kan_prepB<<<(NPAD * Cn) / 256, 256, 0, stream>>>(bw, sw, ss, Bm);
    // 2) single-pass LN stats + pooled + silu/spline A matrix (reads x once: ~310MB)
    kan_pass1<<<Bn, 256, 0, stream>>>(x, lnw, Wc, Bc, A);
    // 3) fused WMMA GEMM: out = A(512x6912) * Bm(1024x6912)^T, N clipped to 1000
    kan_gemm<<<dim3(NPAD / 128, Bn / 64), 256, 0, stream>>>(A, Bm, out);
}
